// AdaptiveGraphConv_83141976916308
// MI455X (gfx1250) — matrix operand order repaired and checked by reading, not compiled, not run on hardware
//
#include <hip/hip_runtime.h>

#define BB    64
#define CIN   64
#define TT    300
#define NJ    25
#define NP    32      // joints padded to 32 for WMMA tiles
#define COUTC 128
#define EMB   32
#define KVP   3
#define TC    4       // timesteps per block in kernel 2
#define SEG   10      // t-segments for attention partials
#define TSEG  (TT / SEG)

typedef __attribute__((ext_vector_type(16))) _Float16 v16h;
typedef __attribute__((ext_vector_type(8)))  _Float16 v8h;
typedef __attribute__((ext_vector_type(8)))  float    v8f;

// ---------------- fragment-layout address helpers (wave32; A/B use contiguous k = 16*(lane>>4) + j) ----
// A tile 16x32 (MxK), 512 halves, lane-major (lane's 16 halves contiguous):
//   lane = m + 16*(k>>4),  half j = k & 15
__device__ __forceinline__ int fragA_off(int m, int k) {
  return (m + 16 * (k >> 4)) * 16 + (k & 15);
}
// B tile 32x16 (KxN), 512 halves, lane-major:
//   lane = n + 16*(k>>4),  half j = k & 15
__device__ __forceinline__ int fragB_off(int k, int n) {
  return (n + 16 * (k >> 4)) * 16 + (k & 15);
}
// C/D 16x16 f32: element v of lane -> row = v + 8*(lane>>4), col = lane&15

__device__ __forceinline__ v16h ldFrag(const _Float16* p) {
  return *reinterpret_cast<const v16h*>(p);
}

// =================== Kernel 1a: embedding + partial attention scores ===================
// Grid (SEG, BB): block (seg,b) accumulates score tile over its 30 timesteps and
// writes a disjoint partial slot (deterministic reduction; no FP atomics).
__global__ __launch_bounds__(256)
void agc_attention_partial_kernel(const float* __restrict__ x,
                                  const float* __restrict__ theta_w,
                                  const float* __restrict__ theta_b,
                                  const float* __restrict__ phi_w,
                                  const float* __restrict__ phi_b,
                                  float* __restrict__ partial) {
  __shared__ __align__(32) _Float16 sWf[8 * 512];   // A-frags of [theta;phi]
  __shared__ __align__(32) _Float16 sXf[4 * 512];   // B-frags of x_t
  __shared__ __align__(32) _Float16 sThA[2 * 512];  // A-frags of th^T (joint x emb)
  __shared__ __align__(32) _Float16 sPhB[2 * 512];  // B-frags of ph   (emb x joint)
  __shared__ float sEb[2 * EMB];

  const int seg  = blockIdx.x;       // 0..SEG-1
  const int b    = blockIdx.y;
  const int tid  = threadIdx.x;
  const int lane = tid & 31;
  const int wave = tid >> 5;
  const int mt   = wave >> 1;        // 0..3 : 16-row tile of combined weights
  const int nt   = wave & 1;         // 0..1 : 16-col tile of padded joints
  const int tA   = mt & 1;           // score A tile (waves 4..7 duplicate 0..3)
  const int tB   = nt;               // score B tile

  // stage combined [theta; phi] weights straight into A-fragment layout
  for (int i = tid; i < 2 * EMB * CIN; i += 256) {
    const int row = i >> 6, c = i & 63;
    const float w = (row < EMB) ? theta_w[row * CIN + c] : phi_w[(row - EMB) * CIN + c];
    sWf[((row >> 4) * 2 + (c >> 5)) * 512 + fragA_off(row & 15, c & 31)] = (_Float16)w;
  }
  if (tid < 2 * EMB) sEb[tid] = (tid < EMB) ? theta_b[tid] : phi_b[tid - EMB];

  // per-thread x-staging bookkeeping hoisted out of the t-loop; tail elements
  // clamp to element 0 (benign duplicate store of the same value).
  int gO[7], lO[7];
#pragma unroll
  for (int j = 0; j < 7; ++j) {
    int i = (j < 6) ? (tid + 256 * j) : (1536 + tid);
    if (i >= CIN * NJ) i = 0;
    const int c = i / NJ, n = i % NJ;
    gO[j] = (b * CIN + c) * TT * NJ + n;
    lO[j] = ((c >> 5) * 2 + (n >> 4)) * 512 + fragB_off(c & 31, n & 15);
  }
  __syncthreads();

  // weight A-fragments resident in VGPRs for the whole segment
  const v16h aw0 = ldFrag(&sWf[(mt * 2 + 0) * 512 + lane * 16]);
  const v16h aw1 = ldFrag(&sWf[(mt * 2 + 1) * 512 + lane * 16]);

  // hoisted D-scatter setup: this thread's 8 D elements are rows rbase..rbase+7
  // at one fixed column -> CONSECUTIVE halves in the destination fragment.
  const int rbase = mt * 16 + 8 * (lane >> 4);
  const int colj  = nt * 16 + (lane & 15);
  float eb[8];
#pragma unroll
  for (int v = 0; v < 8; ++v) eb[v] = sEb[rbase + v];
  _Float16* dstPtr;
  if (mt < 2) {   // th rows -> A-frag of th^T: k = emb row
    dstPtr = &sThA[(colj >> 4) * 512 + ((colj & 15) + 16 * (rbase >> 4)) * 16
                   + (rbase & 15)];
  } else {        // ph rows -> B-frag: k = emb row - EMB
    const int kb = rbase - EMB;
    dstPtr = &sPhB[(colj >> 4) * 512 + ((colj & 15) + 16 * (kb >> 4)) * 16 + (kb & 15)];
  }

  v8f accS = {};  // persistent score accumulator tile
  int t25 = seg * TSEG * NJ;

  for (int t = 0; t < TSEG; ++t) {
    // x slice -> B-fragment layout: 7 unconditional loads + 7 b16 stores
#pragma unroll
    for (int j = 0; j < 7; ++j) sXf[lO[j]] = (_Float16)x[gO[j] + t25];
    __builtin_prefetch(&x[gO[0] + t25 + NJ], 0, 3);   // next slice -> global_prefetch
    t25 += NJ;
    __syncthreads();

    v8f d = {};
    {
      const v16h b0 = ldFrag(&sXf[(0 * 2 + nt) * 512 + lane * 16]);
      d = __builtin_amdgcn_wmma_f32_16x16x32_f16(false, aw0, false, b0, (short)0, d, false, false);
      const v16h b1 = ldFrag(&sXf[(1 * 2 + nt) * 512 + lane * 16]);
      d = __builtin_amdgcn_wmma_f32_16x16x32_f16(false, aw1, false, b1, (short)0, d, false, false);
    }
    // packed scatter: 8 cvt + single ds_store_b128
    v8h h;
#pragma unroll
    for (int v = 0; v < 8; ++v) h[v] = (_Float16)(d[v] + eb[v]);
    *reinterpret_cast<v8h*>(dstPtr) = h;
    __syncthreads();

    // score tile accumulation: one WMMA per wave (K = EMB = 32)
    {
      const v16h aT = ldFrag(&sThA[tA * 512 + lane * 16]);
      const v16h bP = ldFrag(&sPhB[tB * 512 + lane * 16]);
      accS = __builtin_amdgcn_wmma_f32_16x16x32_f16(false, aT, false, bP, (short)0, accS,
                                                    false, false);
    }
  }

  if (wave < 4) {   // waves 4..7 hold duplicate tiles; disjoint partial slot
#pragma unroll
    for (int v = 0; v < 8; ++v) {
      const int n = tA * 16 + v + 8 * (lane >> 4);
      const int m = tB * 16 + (lane & 15);
      if (n < NJ && m < NJ)
        partial[((b * SEG + seg) * NJ + n) * NJ + m] = accS[v];
    }
  }
}

// =================== Kernel 1b: reduce partials + softmax + adjacency ===================
__global__ __launch_bounds__(256)
void agc_softmax_kernel(const float* __restrict__ partial,
                        const float* __restrict__ Bparam, const float* __restrict__ A,
                        float* __restrict__ adj) {
  __shared__ float sScore[NJ][NJ + 1];
  const int b   = blockIdx.x;
  const int tid = threadIdx.x;

  for (int p = tid; p < NJ * NJ; p += 256) {
    float s = 0.f;
#pragma unroll
    for (int g = 0; g < SEG; ++g) s += partial[(b * SEG + g) * NJ * NJ + p];
    sScore[p / NJ][p % NJ] = s;
  }
  __syncthreads();

  if (tid < NJ) {   // row-wise softmax + sum_k(A+B) + K*C
    const int n = tid;
    float mx = -3.0e38f;
    for (int m = 0; m < NJ; ++m) mx = fmaxf(mx, sScore[n][m]);
    float sm = 0.f;
    for (int m = 0; m < NJ; ++m) {
      const float e = __expf(sScore[n][m] - mx);
      sScore[n][m] = e; sm += e;
    }
    const float inv = (float)KVP / sm;
    for (int m = 0; m < NJ; ++m) {
      float base = 0.f;
      for (int k = 0; k < KVP; ++k) {
        const int idx = (k * NJ + n) * NJ + m;
        base += A[idx] + Bparam[idx];
      }
      adj[(b * NJ + n) * NJ + m] = base + sScore[n][m] * inv;
    }
  }
}

// =================== Kernel 2: graph-mix (WMMA) + stacked GEMM + BN + ReLU ==========
// xs = x@adj via WMMA (K = joints, zero-padded); then
// out = relu(BN([w1|wr] @ [xs ; x] + b1 + br)) as one 128x128x128 WMMA GEMM.
__global__ __launch_bounds__(256)
void agc_gemm_kernel(const float* __restrict__ x,  const float* __restrict__ adjg,
                     const float* __restrict__ w1, const float* __restrict__ b1,
                     const float* __restrict__ wr, const float* __restrict__ br,
                     const float* __restrict__ gamma, const float* __restrict__ beta,
                     const float* __restrict__ rmean, const float* __restrict__ rvar,
                     float* __restrict__ out) {
  __shared__ float sXraw[CIN][TC * NJ];              // raw x chunk f32 (25.6 KB)
  __shared__ __align__(32) _Float16 sWf[32 * 512];   // A-frags weights (mt0..7, kt0..3) 32 KB
  __shared__ __align__(32) _Float16 sBf[32 * 512];   // B-frags [xs ; x] (kt0..3, nt0..7) 32 KB
  __shared__ __align__(32) _Float16 sXa[16 * 512];   // A-frags x_t (tl0..3, mt0..3)     16 KB
  __shared__ __align__(32) _Float16 sAdjB[2 * 512];  // B-frags adj (nt0..1)              2 KB
  __shared__ float sAlpha[COUTC];
  __shared__ float sBias[COUTC];

  const int tc   = blockIdx.x;   // 0..74
  const int b    = blockIdx.y;
  const int t0   = tc * TC;
  const int tid  = threadIdx.x;
  const int lane = tid & 31;
  const int wave = tid >> 5;     // = output column tile of the main GEMM

  // stacked weights [w1 | wr] -> A-fragment layout
  for (int i = tid; i < COUTC * 2 * CIN; i += 256) {
    const int o = i >> 7, c = i & 127;
    const float w = (c < CIN) ? w1[o * CIN + c] : wr[o * CIN + (c - CIN)];
    sWf[((o >> 4) * 4 + (c >> 5)) * 512 + fragA_off(o & 15, c & 31)] = (_Float16)w;
  }
  if (tid < COUTC) {   // fold biases + BN into per-channel scale/offset
    const float al = gamma[tid] * rsqrtf(rvar[tid] + 1e-5f);
    sAlpha[tid] = al;
    sBias[tid]  = (b1[tid] + br[tid] - rmean[tid]) * al + beta[tid];
  }
  // adjacency -> B-fragment layout, K rows (joints) zero-padded (WMMA: 0*Inf = NaN!)
  for (int i = tid; i < NP * NP; i += 256) {
    const int n = i >> 5, m = i & 31;
    const float av = (n < NJ && m < NJ) ? adjg[b * NJ * NJ + n * NJ + m] : 0.f;
    sAdjB[(m >> 4) * 512 + fragB_off(n, m & 15)] = (_Float16)av;
  }
  for (int i = tid; i < CIN * TC * NJ; i += 256) {   // 100 contiguous floats/channel
    const int c = i / (TC * NJ), r = i % (TC * NJ);
    sXraw[c][r] = x[((b * CIN + c) * TT + t0) * NJ + r];
  }
  __syncthreads();

  // stage x_t A-frags (M=channel, K=joint zero-padded) and raw-x rows 64..127 of sBf
  for (int i = tid; i < CIN * TC * NP; i += 256) {
    const int c   = i >> 7;       // channel 0..63
    const int col = i & 127;      // tl*32 + joint
    const int tl  = col >> 5, n = col & 31;
    const _Float16 hv = (_Float16)((n < NJ) ? sXraw[c][tl * NJ + n] : 0.f);
    sXa[(tl * 4 + (c >> 4)) * 512 + fragA_off(c & 15, n)] = hv;
    const int krow = CIN + c;     // stacked-K row 64..127
    sBf[((krow >> 5) * 8 + (col >> 4)) * 512 + fragB_off(krow & 31, col & 15)] = hv;
  }
  __syncthreads();

  // graph-mix via WMMA: 32 tiles (tl0..3, mt0..3, nt0..1), 4 per wave;
  // D packed-stored straight into sBf K-rows 0..63 (consecutive halves per thread)
#pragma unroll
  for (int r = 0; r < 4; ++r) {
    const int idx = wave + r * 8;
    const int tl = idx >> 3, mtc = (idx >> 1) & 3, nt2 = idx & 1;
    v8f d = {};
    const v16h a  = ldFrag(&sXa[(tl * 4 + mtc) * 512 + lane * 16]);
    const v16h bb = ldFrag(&sAdjB[nt2 * 512 + lane * 16]);
    d = __builtin_amdgcn_wmma_f32_16x16x32_f16(false, a, false, bb, (short)0, d, false, false);
    const int rb = mtc * 16 + 8 * (lane >> 4);           // K-row base (channel)
    const int cl = tl * 32 + nt2 * 16 + (lane & 15);     // stacked column
    v8h h;
#pragma unroll
    for (int v = 0; v < 8; ++v) h[v] = (_Float16)d[v];
    _Float16* dp = &sBf[((rb >> 5) * 8 + (cl >> 4)) * 512
                        + ((cl & 15) + 16 * (((rb & 31)) >> 4)) * 16 + (rb & 15)];
    *reinterpret_cast<v8h*>(dp) = h;
  }
  __syncthreads();

  // main 128x128x128 GEMM: wave owns column tile `wave`; B-frags hoisted, reused 8x
  v16h bfr[4];
#pragma unroll
  for (int kt = 0; kt < 4; ++kt)
    bfr[kt] = ldFrag(&sBf[(kt * 8 + wave) * 512 + lane * 16]);

  const int col = wave * 16 + (lane & 15);
  const int tl  = col >> 5;
  const int m   = col & 31;

#pragma unroll
  for (int mt = 0; mt < 8; ++mt) {
    v8f d = {};
#pragma unroll
    for (int kt = 0; kt < 4; ++kt) {
      const v16h a = ldFrag(&sWf[(mt * 4 + kt) * 512 + lane * 16]);
      d = __builtin_amdgcn_wmma_f32_16x16x32_f16(false, a, false, bfr[kt], (short)0, d,
                                                 false, false);
    }
#pragma unroll
    for (int v = 0; v < 8; ++v) {
      const int o = mt * 16 + v + 8 * (lane >> 4);
      if (m < NJ) {
        out[((b * COUTC + o) * TT + (t0 + tl)) * NJ + m]
            = fmaxf(0.f, d[v] * sAlpha[o] + sBias[o]);
      }
    }
  }
}

extern "C" void kernel_launch(void* const* d_in, const int* in_sizes, int n_in,
                              void* d_out, int out_size, void* d_ws, size_t ws_size,
                              hipStream_t stream) {
  const float* x  = (const float*)d_in[0];
  const float* tw = (const float*)d_in[1];
  const float* tb = (const float*)d_in[2];
  const float* pw = (const float*)d_in[3];
  const float* pb = (const float*)d_in[4];
  const float* Bp = (const float*)d_in[5];
  const float* A  = (const float*)d_in[6];
  const float* w1 = (const float*)d_in[7];
  const float* b1 = (const float*)d_in[8];
  const float* wr = (const float*)d_in[9];
  const float* br = (const float*)d_in[10];
  const float* gm = (const float*)d_in[11];
  const float* bt = (const float*)d_in[12];
  const float* rm = (const float*)d_in[13];
  const float* rv = (const float*)d_in[14];
  float* out = (float*)d_out;

  float* partial = (float*)d_ws;                       // BB*SEG*625 f32 = 1.6 MB
  float* adj     = partial + (size_t)BB * SEG * NJ * NJ;  // BB*625 f32 = 160 KB

  agc_attention_partial_kernel<<<dim3(SEG, BB), 256, 0, stream>>>(x, tw, tb, pw, pb, partial);
  agc_softmax_kernel<<<BB, 256, 0, stream>>>(partial, Bp, A, adj);
  agc_gemm_kernel<<<dim3(TT / TC, BB), 256, 0, stream>>>(x, adj, w1, b1, wr, br,
                                                         gm, bt, rm, rv, out);
  (void)in_sizes; (void)n_in; (void)out_size; (void)ws_size;
}
